// CRF_403726926304
// MI455X (gfx1250) — compile-verified, matrix operand order checked by reading, not run on hardware
//
#include <hip/hip_runtime.h>
#include <hip/hip_bf16.h>

typedef __attribute__((ext_vector_type(16))) _Float16 v16h;
typedef __attribute__((ext_vector_type(8)))  float    v8f;

#define S_LEN 2048
#define B_SZ  256
#define T_SZ  64
#define ROWS  16     // batch rows per workgroup
#define NTHR  128    // 4 waves

#define LOG2E 1.4426950408889634f
#define LN2   0.6931471805599453f

// LDS-only workgroup barrier: drain DScnt, then split signal/wait.
// Unlike __syncthreads(), does NOT drain LOADcnt, so in-flight global
// prefetch loads stay outstanding across the barrier.
__device__ __forceinline__ void lds_barrier() {
    asm volatile("s_wait_dscnt 0x0" ::: "memory");
    asm volatile("s_barrier_signal -1\n\t"
                 "s_barrier_wait -1" ::: "memory");
}

__global__ void zero_out_kernel(float* out) { out[0] = 0.0f; }

// ---------------- Numerator: gather + reduce per batch element ----------------
__global__ __launch_bounds__(64)
void crf_numerator(const float* __restrict__ em, const int* __restrict__ tags,
                   const unsigned char* __restrict__ mask,
                   const float* __restrict__ startT, const float* __restrict__ endT,
                   const float* __restrict__ trans, float* __restrict__ num)
{
    const int b = blockIdx.x;
    const int t = threadIdx.x;
    __shared__ float red[64];
    __shared__ int   redi[64];

    float acc = 0.0f;
    int   cnt = 0;
    for (int s = t; s < S_LEN; s += 64) {
        int tg           = tags[(size_t)s * B_SZ + b];
        unsigned char mk = mask[(size_t)s * B_SZ + b];
        float e          = em[((size_t)s * B_SZ + b) * T_SZ + tg];
        if (s == 0) {
            acc += startT[tg] + e;
        } else if (mk) {
            int tp = tags[(size_t)(s - 1) * B_SZ + b];
            acc += trans[tp * T_SZ + tg] + e;
        }
        cnt += mk ? 1 : 0;
    }
    red[t] = acc; redi[t] = cnt;
    __syncthreads();
    for (int off = 32; off > 0; off >>= 1) {
        if (t < off) { red[t] += red[t + off]; redi[t] += redi[t + off]; }
        __syncthreads();
    }
    if (t == 0) {
        int seq_end = redi[0] - 1;
        int lt = tags[(size_t)seq_end * B_SZ + b];
        num[b] = red[0] + endT[lt];
    }
}

// ---------------- Forward scan: log-semiring GEMM via WMMA ----------------
// Running score kept in LOG2 units. Per step per wave: one 16x16 F32 tile of
// q = exp2(score2 - rowmax) @ exp(trans), via two v_wmma_f32_16x16x32_f16.
// Raw v_exp_f32 / v_log_f32 (base-2) on the chain; q >= ~0.9 so no denorm
// guards are needed.
__global__ __launch_bounds__(NTHR)
void crf_scan(const float* __restrict__ em, const unsigned char* __restrict__ mask,
              const float* __restrict__ startT, const float* __restrict__ endT,
              const float* __restrict__ trans, const float* __restrict__ num,
              float* __restrict__ out)
{
    __shared__ __align__(16) float    score[ROWS][T_SZ + 8];  // log2 units
    __shared__ __align__(16) float    rowm[ROWS];
    __shared__ __align__(16) _Float16 afrag[2][32][16];       // A frags, ISA layout

    const int tid  = threadIdx.x;
    const int wave = tid >> 5;
    const int lane = tid & 31;
    const int batchBase = blockIdx.x * ROWS;

    // phase-1 role: 128 threads cover the 16x64 score tile, 8 cols each.
    // The 8 threads of a row are consecutive lanes of one wave -> shfl reduce.
    const int p1row = tid >> 3;
    const int p1col = (tid & 7) * 8;

    // WMMA C-layout role: VGPR r, lanes 0-15 -> M=r, lanes 16-31 -> M=r+8; N=lane%16
    const int lg    = lane >> 4;
    const int Nglob = wave * 16 + (lane & 15);
    const int mbase = 8 * lg;

    // ---- Invariant B fragments: E = exp(trans) = 2^(trans*log2e)
    // B 32x16 (f16) layout: lane = N + 16*((kk>>3)&1), slot = 8*(kk>>4)+(kk&7)
    v16h bf0, bf1;
    for (int sl = 0; sl < 16; ++sl) {
        int a = sl >> 3, bb = sl & 7;
        int kk = 16 * a + 8 * lg + bb;
        bf0[sl] = (_Float16)__builtin_amdgcn_exp2f(trans[(kk     ) * T_SZ + Nglob] * LOG2E);
        bf1[sl] = (_Float16)__builtin_amdgcn_exp2f(trans[(kk + 32) * T_SZ + Nglob] * LOG2E);
    }

    // ---- score2_0 = (start_transitions + emissions[0]) * log2e
    for (int c = 0; c < 8; ++c) {
        int col = p1col + c;
        score[p1row][col] = (startT[col] +
            em[((size_t)batchBase + p1row) * T_SZ + col]) * LOG2E;
    }
    __syncthreads();

    // ---- software-pipelined emissions (pre-scaled to log2 units) / mask, s=1
    float emc[8];
    unsigned long long mkc;
    for (int r = 0; r < 8; ++r)
        emc[r] = em[((size_t)1 * B_SZ + batchBase + mbase + r) * T_SZ + Nglob] * LOG2E;
    mkc = *(const unsigned long long*)(mask + (size_t)1 * B_SZ + batchBase + mbase);

    for (int s = 1; s < S_LEN; ++s) {
        // prefetch step s+1; stays in flight across the LDS-only barriers
        float emn[8];
        unsigned long long mkn = 0ull;
        if (s + 1 < S_LEN) {
            for (int r = 0; r < 8; ++r)
                emn[r] = em[((size_t)(s + 1) * B_SZ + batchBase + mbase + r) * T_SZ + Nglob] * LOG2E;
            mkn = *(const unsigned long long*)(mask + (size_t)(s + 1) * B_SZ + batchBase + mbase);
        }

        // phase 1: row max via in-wave shuffle (8-thread row groups, no barrier)
        float4 v0 = *(const float4*)&score[p1row][p1col];
        float4 v1 = *(const float4*)&score[p1row][p1col + 4];
        float mx = fmaxf(fmaxf(fmaxf(v0.x, v0.y), fmaxf(v0.z, v0.w)),
                         fmaxf(fmaxf(v1.x, v1.y), fmaxf(v1.z, v1.w)));
        mx = fmaxf(mx, __shfl_xor(mx, 1, 8));
        mx = fmaxf(mx, __shfl_xor(mx, 2, 8));
        mx = fmaxf(mx, __shfl_xor(mx, 4, 8));
        if ((tid & 7) == 0) rowm[p1row] = mx;

        // p = exp2(score2 - m2), scattered into the ISA A-fragment layout
        // (contiguous per thread -> single ds_store_b128)
        float pv[8] = {v0.x, v0.y, v0.z, v0.w, v1.x, v1.y, v1.z, v1.w};
        for (int c = 0; c < 8; ++c) {
            int K  = p1col + c;
            int h  = K >> 5, kk = K & 31;
            int al = p1row + 16 * ((kk >> 3) & 1);
            int sl = 8 * (kk >> 4) + (kk & 7);
            afrag[h][al][sl] = (_Float16)__builtin_amdgcn_exp2f(pv[c] - mx);
        }
        lds_barrier();

        // WMMA: q(16x16) = p(16x64) @ E(64x16), K split in two halves
        v16h a0 = *(const v16h*)&afrag[0][lane][0];
        v16h a1 = *(const v16h*)&afrag[1][lane][0];
        v8f  c4 = {};
        c4 = __builtin_amdgcn_wmma_f32_16x16x32_f16(false, a0, false, bf0,
                                                    (short)0, c4, false, false);
        c4 = __builtin_amdgcn_wmma_f32_16x16x32_f16(false, a1, false, bf1,
                                                    (short)0, c4, false, false);

        float4 m0 = *(const float4*)&rowm[mbase];
        float4 m1 = *(const float4*)&rowm[mbase + 4];
        float mrow[8] = {m0.x, m0.y, m0.z, m0.w, m1.x, m1.y, m1.z, m1.w};

        // nxt2 = m2 + log2(q) + em2 ; masked update of score (raw v_log_f32)
        for (int r = 0; r < 8; ++r) {
            int M = mbase + r;
            float nxt = mrow[r] + __builtin_amdgcn_logf(c4[r]) + emc[r];
            float old = score[M][Nglob];
            bool mk = ((mkc >> (8 * r)) & 0xFFull) != 0ull;
            score[M][Nglob] = mk ? nxt : old;
        }
        lds_barrier();

        for (int r = 0; r < 8; ++r) emc[r] = emn[r];
        mkc = mkn;
    }

    // ---- denominator per batch row (convert log2 -> ln at the end)
    __syncthreads();
    if (tid < ROWS) {
        const int row = tid;
        float mx = -3.402823466e38f;
        for (int j = 0; j < T_SZ; ++j)
            mx = fmaxf(mx, score[row][j] + endT[j] * LOG2E);
        float sum = 0.0f;
        for (int j = 0; j < T_SZ; ++j)
            sum += __builtin_amdgcn_exp2f(score[row][j] + endT[j] * LOG2E - mx);
        float den = LN2 * (mx + __builtin_amdgcn_logf(sum));
        float llh = num[batchBase + row] - den;
        atomicAdd(out, llh);
    }
}

extern "C" void kernel_launch(void* const* d_in, const int* in_sizes, int n_in,
                              void* d_out, int out_size, void* d_ws, size_t ws_size,
                              hipStream_t stream) {
    const float*         em     = (const float*)d_in[0];
    const int*           tags   = (const int*)d_in[1];
    const unsigned char* mask   = (const unsigned char*)d_in[2];
    const float*         startT = (const float*)d_in[3];
    const float*         endT   = (const float*)d_in[4];
    const float*         trans  = (const float*)d_in[5];
    float* out = (float*)d_out;
    float* num = (float*)d_ws;   // B floats of scratch

    zero_out_kernel<<<1, 1, 0, stream>>>(out);
    crf_numerator<<<B_SZ, 64, 0, stream>>>(em, tags, mask, startT, endT, trans, num);
    crf_scan<<<B_SZ / ROWS, NTHR, 0, stream>>>(em, mask, startT, endT, trans, num, out);
}
